// CvtNodeInitializer_2448131359395
// MI455X (gfx1250) — compile-verified, matrix operand order checked by reading, not run on hardware
//
#include <hip/hip_runtime.h>
#include <stdint.h>

#define H   256
#define K2  512
#define NEGV (-1e30f)

typedef __attribute__((ext_vector_type(16))) _Float16 v16h;
typedef __attribute__((ext_vector_type(8)))  float    v8f;

union V16H { uint4 q[2]; v16h v; };

#if defined(__AMDGCN__)
  #if __has_builtin(__builtin_amdgcn_global_load_async_to_lds_b128)
    #define USE_ASYNC_LDS 1
  #endif
#endif

#if defined(USE_ASYNC_LDS)
typedef int v4i __attribute__((vector_size(16)));
typedef __attribute__((address_space(1))) v4i GlbV4;
typedef __attribute__((address_space(3))) v4i LdsV4;
#endif

__device__ __forceinline__ void wait_async_all() {
#if defined(USE_ASYNC_LDS)
  #if __has_builtin(__builtin_amdgcn_s_wait_asynccnt)
    __builtin_amdgcn_s_wait_asynccnt(0);
  #else
    asm volatile("s_wait_asynccnt 0x0" ::: "memory");
  #endif
#endif
}

__device__ __forceinline__ unsigned fkey(float f) {
  unsigned u = __float_as_uint(f);
  return (u & 0x80000000u) ? ~u : (u | 0x80000000u);
}
__device__ __forceinline__ float fdec(unsigned k) {
  unsigned u = (k & 0x80000000u) ? (k & 0x7fffffffu) : ~k;
  return __uint_as_float(u);
}

// ---- W_msg f32 -> f16, tiled as WbT[kc][n][k]  (kc: K-chunk of 32, n: 0..255, k: 0..31)
__global__ void k_prep_w(const float* __restrict__ W, _Float16* __restrict__ WbT) {
  int t = blockIdx.x * blockDim.x + threadIdx.x;
  if (t >= (K2 * H) / 4) return;
  int idx = t * 4;               // flat index into WbT layout
  int kc  = idx >> 13;           // /8192
  int rem = idx & 8191;
  int n   = rem >> 5;
  int k   = rem & 31;
  const float4 s = *(const float4*)(W + (size_t)n * K2 + kc * 32 + k);
  alignas(8) _Float16 h4[4] = {(_Float16)s.x, (_Float16)s.y, (_Float16)s.z, (_Float16)s.w};
  *(uint2*)(WbT + idx) = *(const uint2*)h4;
}

__global__ void k_init(float* __restrict__ agg, float* __restrict__ denom,
                       unsigned* __restrict__ segmax, int N) {
  int i = blockIdx.x * blockDim.x + threadIdx.x;
  if (i < N * H) agg[i] = 0.f;
  if (i < N) { denom[i] = 0.f; segmax[i] = fkey(NEGV); }
}

// ---- msg = [rel ; node[head]] @ W^T via WMMA f16, + logits = msg @ attn
// 64-edge M tile, full 256-col N, K=512 in 16 chunks of 32.
// 8 waves: wave w -> M block (w & 3), N half ((w>>2)*128) -> 8 WMMAs/chunk.
__launch_bounds__(256)
__global__ void k_msg_gemm(const float* __restrict__ rel,
                           const float* __restrict__ node,
                           const _Float16* __restrict__ WbT,
                           const float* __restrict__ attn,
                           const long long* __restrict__ eidx,
                           float* __restrict__ msg,
                           float* __restrict__ logits,
                           int E) {
  __shared__ _Float16 Alds[2][64 * 40];    // 64 edge rows x 32 K (padded to 40), double buffered
  __shared__ _Float16 Blds[2][256 * 40];   // 256 cols x 32 K (padded to 40), double buffered
  __shared__ float    Llds[64];
  __shared__ int      Hlds[64];

  const int tid = threadIdx.x;
  const int e0  = blockIdx.x * 64;

  if (tid < 64) {
    int e = e0 + tid;
    Hlds[tid] = (e < E) ? (int)eidx[e] : 0;   // heads
    Llds[tid] = 0.f;
  }
  __syncthreads();                            // Hlds needed by A staging

  const int lane = tid & 31;
  const int wave = tid >> 5;
  const int half = lane >> 4;
  const int l15  = lane & 15;
  const int mblk = wave & 3;            // 16-row M block within the 64-edge tile
  const int n0   = (wave >> 2) * 128;   // 128-col half per wave group

  const int arow = tid >> 2;            // 0..63
  const int akq  = (tid & 3) * 8;       // 0,8,16,24
  const int ae   = e0 + arow;

  auto stageA = [&](int kc, int buf) {
    int kg = kc * 32 + akq;
    float4 s0, s1;
    if (ae < E) {
      const float* base = (kg < H) ? (rel  + (size_t)ae * H + kg)
                                   : (node + (size_t)Hlds[arow] * H + (kg - H));
      s0 = *(const float4*)base;
      s1 = *(const float4*)(base + 4);
    } else {
      s0 = make_float4(0.f, 0.f, 0.f, 0.f);
      s1 = s0;
    }
    alignas(16) _Float16 h8[8] = {(_Float16)s0.x, (_Float16)s0.y, (_Float16)s0.z, (_Float16)s0.w,
                                  (_Float16)s1.x, (_Float16)s1.y, (_Float16)s1.z, (_Float16)s1.w};
    *(uint4*)(&Alds[buf][arow * 40 + akq]) = *(const uint4*)h8;
  };

  auto stageB = [&](int kc, int buf) {
    const _Float16* src = WbT + (size_t)kc * 8192 + tid * 32;   // one 64B row per thread
    _Float16* dst = &Blds[buf][tid * 40];
#if defined(USE_ASYNC_LDS)
    __builtin_amdgcn_global_load_async_to_lds_b128((GlbV4*)src, (LdsV4*)dst, 0,  0);
    __builtin_amdgcn_global_load_async_to_lds_b128((GlbV4*)src, (LdsV4*)dst, 16, 0);
    __builtin_amdgcn_global_load_async_to_lds_b128((GlbV4*)src, (LdsV4*)dst, 32, 0);
    __builtin_amdgcn_global_load_async_to_lds_b128((GlbV4*)src, (LdsV4*)dst, 48, 0);
#else
    const uint4* s4 = (const uint4*)src;
    uint4* d4 = (uint4*)dst;
    d4[0] = s4[0]; d4[1] = s4[1]; d4[2] = s4[2]; d4[3] = s4[3];
#endif
  };

  v8f acc[8] = {v8f{}, v8f{}, v8f{}, v8f{}, v8f{}, v8f{}, v8f{}, v8f{}};

  // prologue: stage chunk 0 into buffer 0
  stageB(0, 0);
  stageA(0, 0);
  wait_async_all();
  __syncthreads();

  for (int kc = 0; kc < 16; ++kc) {
    const int buf = kc & 1;
    if (kc < 15) {               // overlap next-chunk staging with this chunk's WMMAs
      stageB(kc + 1, 1 - buf);
      stageA(kc + 1, 1 - buf);
    }

    V16H a;
    {
      const int ab = (mblk * 16 + l15) * 40 + half * 8;
      a.q[0] = *(const uint4*)(&Alds[buf][ab]);
      a.q[1] = *(const uint4*)(&Alds[buf][ab + 16]);
    }
#pragma unroll
    for (int t = 0; t < 8; ++t) {
      V16H b;
      const int nb = (n0 + t * 16 + l15) * 40 + half * 8;
      b.q[0] = *(const uint4*)(&Blds[buf][nb]);
      b.q[1] = *(const uint4*)(&Blds[buf][nb + 16]);
      acc[t] = __builtin_amdgcn_wmma_f32_16x16x32_f16(
          false, a.v, false, b.v, (short)0, acc[t], false, false);
    }

    if (kc < 15) {
      wait_async_all();
      __syncthreads();
    }
  }

  // epilogue: store msg (f32) + reduce logits via LDS float atomics
  float lp[8] = {0.f, 0.f, 0.f, 0.f, 0.f, 0.f, 0.f, 0.f};
#pragma unroll
  for (int t = 0; t < 8; ++t) {
    int n = n0 + t * 16 + l15;
    float av = attn[n];
#pragma unroll
    for (int i = 0; i < 8; ++i) {
      int er = e0 + mblk * 16 + half * 8 + i;
      float v = acc[t][i];
      if (er < E) msg[(size_t)er * H + n] = v;
      lp[i] += v * av;
    }
  }
#pragma unroll
  for (int i = 0; i < 8; ++i)
    atomicAdd(&Llds[mblk * 16 + half * 8 + i], lp[i]);
  __syncthreads();
  if (tid < 64 && (e0 + tid) < E) logits[e0 + tid] = Llds[tid];
}

__global__ void k_segmax(const float* __restrict__ logits, const long long* __restrict__ eidx,
                         const int* __restrict__ is_cvt, unsigned* __restrict__ segmax, int E) {
  int e = blockIdx.x * blockDim.x + threadIdx.x;
  if (e >= E) return;
  int tail = (int)eidx[(size_t)E + e];
  float ml = is_cvt[tail] ? logits[e] : NEGV;
  atomicMax(segmax + tail, fkey(ml));
}

__global__ void k_expdenom(const float* __restrict__ logits, const long long* __restrict__ eidx,
                           const int* __restrict__ is_cvt, const unsigned* __restrict__ segmax,
                           float* __restrict__ evalb, float* __restrict__ denom, int E) {
  int e = blockIdx.x * blockDim.x + threadIdx.x;
  if (e >= E) return;
  int tail = (int)eidx[(size_t)E + e];
  float ev = 0.f;
  if (is_cvt[tail]) {
    float sm = fdec(segmax[tail]);
    if (sm <= NEGV) sm = 0.f;
    ev = __expf(logits[e] - sm);
  }
  evalb[e] = ev;
  if (ev != 0.f) atomicAdd(denom + tail, ev);
}

__global__ void k_agg(const float* __restrict__ msg, const float* __restrict__ evalb,
                      const float* __restrict__ denom, const long long* __restrict__ eidx,
                      float* __restrict__ agg, int E) {
  int e = blockIdx.x;
  if (e >= E) return;
  float ev = evalb[e];
  if (ev == 0.f) return;
  int tail = (int)eidx[(size_t)E + e];
  float d = denom[tail];
  float w = ev / (d > 0.f ? d : 1.f);
  int h = threadIdx.x;
  atomicAdd(agg + (size_t)tail * H + h, msg[(size_t)e * H + h] * w);
}

__global__ void k_final(const float* __restrict__ node, const float* __restrict__ agg,
                        const float* __restrict__ shc, const int* __restrict__ is_cvt,
                        float* __restrict__ out, int N) {
  int i = blockIdx.x * blockDim.x + threadIdx.x;
  if (i >= N * H) return;
  int n = i >> 8, h = i & 255;
  out[i] = is_cvt[n] ? (agg[i] + shc[h]) : node[i];
}

extern "C" void kernel_launch(void* const* d_in, const int* in_sizes, int n_in,
                              void* d_out, int out_size, void* d_ws, size_t ws_size,
                              hipStream_t stream) {
  const float*     node = (const float*)d_in[0];
  const float*     rel  = (const float*)d_in[1];
  const float*     Wm   = (const float*)d_in[2];
  const float*     shc  = (const float*)d_in[3];
  const float*     attn = (const float*)d_in[4];
  const long long* eidx = (const long long*)d_in[5];
  const int*       cvt  = (const int*)d_in[6];
  const int N = in_sizes[0] / H;
  const int E = in_sizes[1] / H;

  char* ws = (char*)d_ws;
  size_t off = 0;
  auto take = [&](size_t bytes) -> void* {
    void* p = ws + off;
    off = (off + bytes + 255) & ~(size_t)255;
    return p;
  };
  _Float16* WbT    = (_Float16*)take((size_t)K2 * H * sizeof(_Float16));
  float*    msg    = (float*)   take((size_t)E * H * sizeof(float));
  float*    logits = (float*)   take((size_t)E * sizeof(float));
  float*    evalb  = (float*)   take((size_t)E * sizeof(float));
  unsigned* segmax = (unsigned*)take((size_t)N * sizeof(unsigned));
  float*    denom  = (float*)   take((size_t)N * sizeof(float));
  float*    agg    = (float*)   take((size_t)N * H * sizeof(float));

  k_prep_w<<<((K2 * H) / 4 + 255) / 256, 256, 0, stream>>>(Wm, WbT);
  k_init<<<((size_t)N * H + 255) / 256, 256, 0, stream>>>(agg, denom, segmax, N);
  k_msg_gemm<<<(E + 63) / 64, 256, 0, stream>>>(rel, node, WbT, attn, eidx, msg, logits, E);
  k_segmax<<<(E + 255) / 256, 256, 0, stream>>>(logits, eidx, cvt, segmax, E);
  k_expdenom<<<(E + 255) / 256, 256, 0, stream>>>(logits, eidx, cvt, segmax, evalb, denom, E);
  k_agg<<<E, H, 0, stream>>>(msg, evalb, denom, eidx, agg, E);
  k_final<<<((size_t)N * H + 255) / 256, 256, 0, stream>>>(node, agg, shc, cvt, (float*)d_out, N);
}
